// Mambaspe_4020089389062
// MI455X (gfx1250) — matrix operand order repaired and checked by reading, not compiled, not run on hardware
//
#include <hip/hip_runtime.h>
#include <hip/hip_bf16.h>

// ---------------------------------------------------------------------------
// MI455X (gfx1250) implementation of the Mamba-style SS2D block.
//
// Roofline: ~10 GFLOP vs ~150-200 MB HBM traffic @ 23.3 TB/s -> bandwidth
// bound. Strategy: bf16 intermediates for the two big tensors, bf16 WMMA
// (v_wmma_f32_16x16x32_bf16, f32 accumulate) for the two 32768x256x256 GEMMs,
// wave-per-(pixel,dir) VALU scan (L=8, N=16 state fully in registers).
// Kernel-5 A staging uses GLOBAL_LOAD_ASYNC_TO_LDS_B128 (ASYNCcnt path).
//
// NOTE: reference's z = softmax(..., axis=1) over a size-1 axis == 1.0
// exactly, so the skip conv branch is a mathematical no-op and is skipped.
//
// Workspace layout (assumes ws_size >= 128 MiB):
//   [0,          16 MiB)  xi    bf16 (B, Din, HW)        in_proj output
//   [16 MiB,     48 MiB)  u     f32  (B*HW, CN, G)       conv+silu output
//   [48 MiB,    112 MiB)  ypart f32  (B*HW, K, CN, L)    scan output
//   [112 MiB,   128 MiB)  yln   bf16 (B*HW, Din)         LN output (GEMM2 A)
// ---------------------------------------------------------------------------

typedef __attribute__((ext_vector_type(16))) __bf16 v16bf;
typedef __attribute__((ext_vector_type(8)))  __bf16 v8bf;
typedef __attribute__((ext_vector_type(2)))  __bf16 v2bf;
typedef __attribute__((ext_vector_type(8)))  float  v8f;

#define HW    4096
#define CIN   256
#define DIN   256
#define CN    32
#define NG    8      // groups == sequence length L
#define NS    16     // SSM state size

// ===========================================================================
// Kernel 1: in_proj GEMM.  xi[b,d,hw] = sum_c x[b,c,hw] * w[d,c]  (bf16 out)
// A (M=32768 x K=256) comes from NCHW x (transposed load), B = in_proj_w.
// Block tile: 128m x 64n, 8 waves, each wave 32m x 32n (4 WMMA accumulators).
// ===========================================================================
__global__ void __launch_bounds__(256)
k_inproj_gemm(const float* __restrict__ x, const float* __restrict__ w,
              __bf16* __restrict__ xi)
{
    __shared__ __attribute__((aligned(16))) __bf16 Asm[128 * 32]; // [m][k]
    __shared__ __attribute__((aligned(16))) __bf16 Bsm[64 * 32];  // [n][k]

    const int t   = threadIdx.x;
    const int m0  = blockIdx.x * 128;          // pixel tile base
    const int b   = m0 >> 12;                  // 128 | 4096, never crosses batch
    const int hw0 = m0 & (HW - 1);
    const int n0  = blockIdx.y * 64;           // output-channel tile base

    const int wave = t >> 5, lane = t & 31;
    const int mbase = (wave & 3) * 32;
    const int nbase = (wave >> 2) * 32;
    const int row = lane & 15;
    const int hi  = lane >> 4;                 // 0 | 1 (half-wave)
    const int ka  = hi * 8;                    // A frag K offset in 16-group
    const int kb  = hi * 8;                    // B frag K offset

    v8f acc[2][2] = {};

    // staging-load index maps: each thread owns a 2k x 8m micro-tile so the
    // transpose lands as packed ds_store_b32 (2 adjacent bf16 along K).
    const int akp   = (t >> 4) * 2;            // 0,2,..,30  A k pair
    const int amseg = (t & 15) * 8;            // A m segment (8 f32)
    const int bn    = t & 63;                  // B n row
    const int bkh   = (t >> 6) * 8;            // B k half-row (8 f32)

    for (int kk = 0; kk < CIN; kk += 32) {
        // ---- stage A tile: 128m x 32k (transpose NCHW -> [m][k] bf16) ----
        {
            const float* src0 = x + ((size_t)(b * CIN + kk + akp)) * HW + hw0 + amseg;
            const float* src1 = src0 + HW;
            if (kk + 32 < CIN)
                __builtin_prefetch(src0 + 32 * HW, 0, 1);   // global_prefetch_b8
            float4 a0 = *(const float4*)(src0);
            float4 a1 = *(const float4*)(src0 + 4);
            float4 b0 = *(const float4*)(src1);
            float4 b1 = *(const float4*)(src1 + 4);
            const float r0[8] = {a0.x, a0.y, a0.z, a0.w, a1.x, a1.y, a1.z, a1.w};
            const float r1[8] = {b0.x, b0.y, b0.z, b0.w, b1.x, b1.y, b1.z, b1.w};
            #pragma unroll
            for (int j = 0; j < 8; ++j) {
                v2bf pr;
                pr[0] = (__bf16)r0[j];          // K = akp
                pr[1] = (__bf16)r1[j];          // K = akp+1
                *(v2bf*)&Asm[(amseg + j) * 32 + akp] = pr;   // ds_store_b32
            }
        }
        // ---- stage B tile: 64n x 32k ([n][k] bf16, memory-order rows) ----
        {
            const float* ws = w + (size_t)(n0 + bn) * CIN + kk + bkh;
            float4 v0 = *(const float4*)(ws);
            float4 v1 = *(const float4*)(ws + 4);
            v8bf pk;
            pk[0] = (__bf16)v0.x; pk[1] = (__bf16)v0.y;
            pk[2] = (__bf16)v0.z; pk[3] = (__bf16)v0.w;
            pk[4] = (__bf16)v1.x; pk[5] = (__bf16)v1.y;
            pk[6] = (__bf16)v1.z; pk[7] = (__bf16)v1.w;
            *(v8bf*)&Bsm[bn * 32 + bkh] = pk;
        }
        __syncthreads();

        // ---- fragments per ISA 16-bit A 16x32 / B 32x16 layouts ----
        v16bf afr[2], bfr[2];
        #pragma unroll
        for (int i = 0; i < 2; ++i) {
            const __bf16* ap = &Asm[(mbase + i * 16 + row) * 32 + ka];
            v8bf lo = *(const v8bf*)ap;            // K = ka .. ka+7
            v8bf hh = *(const v8bf*)(ap + 16);     // K = 16+ka .. 16+ka+7
            #pragma unroll
            for (int q = 0; q < 8; ++q) { afr[i][q] = lo[q]; afr[i][8 + q] = hh[q]; }
        }
        #pragma unroll
        for (int j = 0; j < 2; ++j) {
            const __bf16* bp = &Bsm[(nbase + j * 16 + row) * 32 + kb];
            v8bf lo = *(const v8bf*)bp;            // K = kb .. kb+7
            v8bf hh = *(const v8bf*)(bp + 16);     // K = 16+kb .. 16+kb+7
            #pragma unroll
            for (int q = 0; q < 8; ++q) { bfr[j][q] = lo[q]; bfr[j][8 + q] = hh[q]; }
        }
        #pragma unroll
        for (int i = 0; i < 2; ++i)
            #pragma unroll
            for (int j = 0; j < 2; ++j)
                acc[i][j] = __builtin_amdgcn_wmma_f32_16x16x32_bf16(
                    false, afr[i], false, bfr[j], (short)0, acc[i][j],
                    false, false);
        __syncthreads();
    }

    // ---- store C (VGPR r: lanes0-15 M=r, lanes16-31 M=8+r; N=lane&15) ----
    #pragma unroll
    for (int i = 0; i < 2; ++i)
        #pragma unroll
        for (int j = 0; j < 2; ++j)
            #pragma unroll
            for (int r = 0; r < 8; ++r) {
                int ml = mbase + i * 16 + r + hi * 8;
                int d  = n0 + nbase + j * 16 + row;
                xi[((size_t)(b * DIN + d)) * HW + (hw0 + ml)] = (__bf16)acc[i][j][r];
            }
}

// ===========================================================================
// Kernel 2: depthwise 3x3 conv (pad 1) + bias + SiLU, with channel
// re-layout d = c*G+g  ->  u[(pixel*CN + c)*G + g]  (scan-friendly).
// ===========================================================================
__global__ void __launch_bounds__(256)
k_dwconv_silu(const __bf16* __restrict__ xi, const float* __restrict__ cw,
              const float* __restrict__ cb, float* __restrict__ u)
{
    const int idx = blockIdx.x * 256 + threadIdx.x;     // B*DIN*HW threads
    const int hw = idx & (HW - 1);
    const int d  = (idx >> 12) & 255;
    const int b  = idx >> 20;
    const int h = hw >> 6, w = hw & 63;

    const __bf16* base = xi + ((size_t)(b * DIN + d) << 12);
    float acc = cb[d];
    #pragma unroll
    for (int dy = -1; dy <= 1; ++dy)
        #pragma unroll
        for (int dx = -1; dx <= 1; ++dx) {
            int hh = h + dy, ww = w + dx;
            if ((unsigned)hh < 64u && (unsigned)ww < 64u)
                acc += (float)base[hh * 64 + ww] * cw[d * 9 + (dy + 1) * 3 + (dx + 1)];
        }
    float s = acc / (1.f + __expf(-acc));               // SiLU
    int p = (b << 12) | hw;
    u[((size_t)p * CN + (d >> 3)) * NG + (d & 7)] = s;
}

// ===========================================================================
// Kernel 3: bidirectional selective scan. One wave per (pixel, k).
// lane = channel d (CN=32). L=8 sequence, N=16 state in registers.
//   x_dbl = x_proj_w[k] (34x32) @ u_k (32x8)  (rows: 2 dt | 16 B | 16 C)
//   delta = softplus(dt_proj_w @ dts + b);  h-scan;  y += Ds*u.
// ===========================================================================
__global__ void __launch_bounds__(256)
k_scan(const float* __restrict__ u,   const float* __restrict__ xpw,
       const float* __restrict__ dtw, const float* __restrict__ dtb,
       const float* __restrict__ alog,const float* __restrict__ dsv,
       float* __restrict__ ypart)
{
    __shared__ float Ush[8][CN][NG];
    __shared__ float Dts[8][2][NG];
    __shared__ float Bvs[8][NS][NG];
    __shared__ float Cvs[8][NS][NG];

    const int t = threadIdx.x, w = t >> 5, lane = t & 31;
    const int wid   = blockIdx.x * 8 + w;
    const int pixel = wid >> 1;
    const int k     = wid & 1;
    const int d     = lane;

    // sequence (reversed for k=1 == xs[:,1] = u[...,::-1])
    const float* ub = u + ((size_t)pixel * CN + d) * NG;
    float ul[NG];
    #pragma unroll
    for (int l = 0; l < NG; ++l) ul[l] = ub[k ? (7 - l) : l];
    #pragma unroll
    for (int l = 0; l < NG; ++l) Ush[w][d][l] = ul[l];
    __syncthreads();

    // ---- x_dbl: lane computes row c'=lane (lanes 0,1 also rows 32,33) ----
    {
        float rs[NG] = {};
        const float* pw = xpw + ((size_t)k * 34 + lane) * CN;
        for (int dd = 0; dd < CN; ++dd) {
            float wv = pw[dd];
            #pragma unroll
            for (int l = 0; l < NG; ++l) rs[l] += wv * Ush[w][dd][l];
        }
        if (lane < 2) {
            #pragma unroll
            for (int l = 0; l < NG; ++l) Dts[w][lane][l] = rs[l];
        } else if (lane < 18) {
            #pragma unroll
            for (int l = 0; l < NG; ++l) Bvs[w][lane - 2][l] = rs[l];
        } else {
            #pragma unroll
            for (int l = 0; l < NG; ++l) Cvs[w][lane - 18][l] = rs[l];
        }
        if (lane < 2) {                       // rows 32,33 -> Cv rows 14,15
            float rs2[NG] = {};
            const float* pw2 = xpw + ((size_t)k * 34 + 32 + lane) * CN;
            for (int dd = 0; dd < CN; ++dd) {
                float wv = pw2[dd];
                #pragma unroll
                for (int l = 0; l < NG; ++l) rs2[l] += wv * Ush[w][dd][l];
            }
            #pragma unroll
            for (int l = 0; l < NG; ++l) Cvs[w][14 + lane][l] = rs2[l];
        }
    }
    __syncthreads();

    // ---- delta = softplus(dts @ dt_proj_w[k,d,:] + b[k,d]) ----
    const float w0 = dtw[(k * CN + d) * 2 + 0];
    const float w1 = dtw[(k * CN + d) * 2 + 1];
    const float bb = dtb[k * CN + d];
    float dt[NG];
    #pragma unroll
    for (int l = 0; l < NG; ++l) {
        float r = Dts[w][0][l] * w0 + Dts[w][1][l] * w1 + bb;
        dt[l] = (r > 20.f) ? r : log1pf(__expf(r));
    }

    float A[NS];
    #pragma unroll
    for (int n = 0; n < NS; ++n)
        A[n] = -__expf(alog[((size_t)k * CN + d) * NS + n]);
    const float Dv = dsv[k * CN + d];

    float h[NS] = {};
    float ys[NG];
    #pragma unroll
    for (int l = 0; l < NG; ++l) {
        const float dtl = dt[l];
        const float du  = dtl * ul[l];
        float y = 0.f;
        #pragma unroll
        for (int n = 0; n < NS; ++n) {
            h[n] = h[n] * __expf(dtl * A[n]) + du * Bvs[w][n][l];
            y += h[n] * Cvs[w][n][l];
        }
        ys[l] = y + Dv * ul[l];
    }
    float* yp = ypart + (((size_t)pixel * 2 + k) * CN + d) * NG;
    #pragma unroll
    for (int l = 0; l < NG; ++l) yp[l] = ys[l];
}

// ===========================================================================
// Kernel 4: combine fwd + reversed bwd, LayerNorm over CN, re-interleave
// channels to din = g*CN + c, write bf16 (row-major A for out_proj GEMM).
// z multiplier == 1 exactly (softmax over singleton) -> omitted.
// ===========================================================================
__global__ void __launch_bounds__(256)
k_ln(const float* __restrict__ ypart, const float* __restrict__ g,
     const float* __restrict__ bta, __bf16* __restrict__ yln)
{
    const int idx = blockIdx.x * 256 + threadIdx.x;     // B*HW*G threads
    const int p  = idx >> 3;
    const int gg = idx & 7;
    const float* y0 = ypart + ((size_t)p * 2 + 0) * (CN * NG);
    const float* y1 = ypart + ((size_t)p * 2 + 1) * (CN * NG);

    float v[CN];
    float mu = 0.f;
    #pragma unroll
    for (int d = 0; d < CN; ++d) {
        float a = y0[d * NG + gg] + y1[d * NG + (7 - gg)];
        v[d] = a; mu += a;
    }
    mu *= (1.f / CN);
    float var = 0.f;
    #pragma unroll
    for (int d = 0; d < CN; ++d) { float c = v[d] - mu; var += c * c; }
    var *= (1.f / CN);
    const float rstd = rsqrtf(var + 1e-5f);

    __bf16* o = yln + (size_t)p * DIN + gg * CN;
    #pragma unroll
    for (int d = 0; d < CN; ++d)
        o[d] = (__bf16)((v[d] - mu) * rstd * g[d] + bta[d]);
}

// ===========================================================================
// Kernel 5: out_proj GEMM. out[b,co,hw] = sum_din yln[m,din]*w[co,din] (f32)
// Same WMMA core as kernel 1. A is bf16 row-major and is staged into LDS
// with GLOBAL_LOAD_ASYNC_TO_LDS_B128 (pure copy -> async DMA path, no VGPR
// round trip; tracked by ASYNCcnt, drained with s_wait_asynccnt 0).
// ===========================================================================
__global__ void __launch_bounds__(256)
k_outproj_gemm(const __bf16* __restrict__ a, const float* __restrict__ w,
               float* __restrict__ out)
{
    __shared__ __attribute__((aligned(16))) __bf16 Asm[128 * 32];
    __shared__ __attribute__((aligned(16))) __bf16 Bsm[64 * 32];

    const int t   = threadIdx.x;
    const int m0  = blockIdx.x * 128;
    const int b   = m0 >> 12;
    const int hw0 = m0 & (HW - 1);
    const int n0  = blockIdx.y * 64;

    const int wave = t >> 5, lane = t & 31;
    const int mbase = (wave & 3) * 32;
    const int nbase = (wave >> 2) * 32;
    const int row = lane & 15;
    const int hi  = lane >> 4;
    const int ka  = hi * 8;
    const int kb  = hi * 8;

    v8f acc[2][2] = {};

    const int am  = t >> 1;                    // 0..127
    const int akh = (t & 1) * 16;              // 0 | 16 (bf16 elements)
    const int bn  = t & 63;
    const int bkh = (t >> 6) * 8;

    // LDS byte address of this thread's A-staging slot (low 32 bits of the
    // flat shared address are the LDS offset per the aperture rules).
    const unsigned lds_dst = (unsigned)(unsigned long long)&Asm[am * 32 + akh];
    const unsigned long long abase = (unsigned long long)a;

    for (int kk = 0; kk < DIN; kk += 32) {
        // ---- stage A: async DMA bf16 global -> LDS (32 B per thread) ----
        {
            unsigned voff = ((unsigned)((m0 + am) * DIN + kk + akh)) * 2u;
            // INST_OFFSET is added to BOTH the global and LDS addresses, so
            // offset:16 covers the second 16-byte half of the 32-byte copy.
            asm volatile(
                "global_load_async_to_lds_b128 %0, %1, %2\n\t"
                "global_load_async_to_lds_b128 %0, %1, %2 offset:16"
                :
                : "v"(lds_dst), "v"(voff), "s"(abase)
                : "memory");
        }
        // ---- stage B: f32 -> bf16 ----
        {
            const float* ws = w + (size_t)(n0 + bn) * DIN + kk + bkh;
            float4 v0 = *(const float4*)(ws);
            float4 v1 = *(const float4*)(ws + 4);
            v8bf pk;
            pk[0] = (__bf16)v0.x; pk[1] = (__bf16)v0.y;
            pk[2] = (__bf16)v0.z; pk[3] = (__bf16)v0.w;
            pk[4] = (__bf16)v1.x; pk[5] = (__bf16)v1.y;
            pk[6] = (__bf16)v1.z; pk[7] = (__bf16)v1.w;
            *(v8bf*)&Bsm[bn * 32 + bkh] = pk;
        }
        asm volatile("s_wait_asynccnt 0x0" ::: "memory");   // drain async DMA
        __syncthreads();

        v16bf afr[2], bfr[2];
        #pragma unroll
        for (int i = 0; i < 2; ++i) {
            const __bf16* ap = &Asm[(mbase + i * 16 + row) * 32 + ka];
            v8bf lo = *(const v8bf*)ap;
            v8bf hh = *(const v8bf*)(ap + 16);
            #pragma unroll
            for (int q = 0; q < 8; ++q) { afr[i][q] = lo[q]; afr[i][8 + q] = hh[q]; }
        }
        #pragma unroll
        for (int j = 0; j < 2; ++j) {
            const __bf16* bp = &Bsm[(nbase + j * 16 + row) * 32 + kb];
            v8bf lo = *(const v8bf*)bp;
            v8bf hh = *(const v8bf*)(bp + 16);
            #pragma unroll
            for (int q = 0; q < 8; ++q) { bfr[j][q] = lo[q]; bfr[j][8 + q] = hh[q]; }
        }
        #pragma unroll
        for (int i = 0; i < 2; ++i)
            #pragma unroll
            for (int j = 0; j < 2; ++j)
                acc[i][j] = __builtin_amdgcn_wmma_f32_16x16x32_bf16(
                    false, afr[i], false, bfr[j], (short)0, acc[i][j],
                    false, false);
        __syncthreads();
    }

    #pragma unroll
    for (int i = 0; i < 2; ++i)
        #pragma unroll
        for (int j = 0; j < 2; ++j)
            #pragma unroll
            for (int r = 0; r < 8; ++r) {
                int ml = mbase + i * 16 + r + hi * 8;
                int co = n0 + nbase + j * 16 + row;
                out[((size_t)(b * CIN + co)) * HW + (hw0 + ml)] = acc[i][j][r];
            }
}

// ===========================================================================
extern "C" void kernel_launch(void* const* d_in, const int* in_sizes, int n_in,
                              void* d_out, int out_size, void* d_ws, size_t ws_size,
                              hipStream_t stream)
{
    const float* x    = (const float*)d_in[0];
    const float* w_in = (const float*)d_in[1];
    // d_in[2] skip_w, d_in[3] skip_b: dead (softmax over singleton axis == 1)
    const float* cw   = (const float*)d_in[4];
    const float* cb   = (const float*)d_in[5];
    const float* xpw  = (const float*)d_in[6];
    const float* dtw  = (const float*)d_in[7];
    const float* dtbp = (const float*)d_in[8];
    const float* alog = (const float*)d_in[9];
    const float* dsv  = (const float*)d_in[10];
    const float* lng  = (const float*)d_in[11];
    const float* lnb  = (const float*)d_in[12];
    const float* w_out= (const float*)d_in[13];
    float* out = (float*)d_out;

    char* ws = (char*)d_ws;
    __bf16* xi    = (__bf16*)(ws);                         //  16 MiB
    float*  u     = (float*)(ws + (16u << 20));            //  32 MiB
    float*  ypart = (float*)(ws + (48u << 20));            //  64 MiB
    __bf16* yln   = (__bf16*)(ws + (112u << 20));          //  16 MiB

    dim3 gemm_grid(256, 4);   // M/128 x N/64
    k_inproj_gemm <<<gemm_grid, 256, 0, stream>>>(x, w_in, xi);
    k_dwconv_silu <<<32768,     256, 0, stream>>>(xi, cw, cb, u);
    k_scan        <<<8192,      256, 0, stream>>>(u, xpw, dtw, dtbp, alog, dsv, ypart);
    k_ln          <<<1024,      256, 0, stream>>>(ypart, lng, lnb, yln);
    k_outproj_gemm<<<gemm_grid, 256, 0, stream>>>(yln, w_out, out);
}
